// MotifResponseLayer_10874857193549
// MI455X (gfx1250) — compile-verified, matrix operand order checked by reading, not run on hardware
//
#include <hip/hip_runtime.h>
#include <hip/hip_bf16.h>
#include <math.h>

// ---------------- problem constants (match reference) ----------------
#define N_NODES  16384
#define D_IN     256
#define D_OUT    256
#define HID      256
#define F_       8
#define L_       4
#define GAMMA    0.5f
#define M_TRI    200000
#define M_CL4    200000
#define IN_CH    272          // 2*F_ + D_IN
#define K1_PAD   288          // IN_CH padded to multiple of 32
#define ROWS_PER_BLOCK 64

typedef __attribute__((ext_vector_type(16))) __bf16 v16bf;
typedef __attribute__((ext_vector_type(8)))  float  v8f;

// ---------------- small helpers ----------------
__device__ __forceinline__ float softplus_f(float x) {
    return (x > 20.f) ? x : log1pf(expf(x));
}

template <int S>
__device__ __forceinline__ void row_norm(float a[S][S]) {
    #pragma unroll
    for (int i = 0; i < S; ++i) {
        float s = 0.f;
        #pragma unroll
        for (int j = 0; j < S; ++j) s += a[i][j];
        float inv = 1.f / fmaxf(s, 1e-12f);
        #pragma unroll
        for (int j = 0; j < S; ++j) a[i][j] *= inv;
    }
}

// traces of A^1..A^4 for a tiny SxS matrix
template <int S>
__device__ __forceinline__ void power_traces(const float a[S][S], float tr[4]) {
    float b[S][S];
    float t1 = 0.f;
    #pragma unroll
    for (int i = 0; i < S; ++i) t1 += a[i][i];
    #pragma unroll
    for (int i = 0; i < S; ++i)
        #pragma unroll
        for (int j = 0; j < S; ++j) {
            float s = 0.f;
            #pragma unroll
            for (int k = 0; k < S; ++k) s += a[i][k] * a[k][j];
            b[i][j] = s;
        }
    float t2 = 0.f, t3 = 0.f, t4 = 0.f;
    #pragma unroll
    for (int i = 0; i < S; ++i) {
        t2 += b[i][i];
        #pragma unroll
        for (int j = 0; j < S; ++j) {
            t3 += a[i][j] * b[j][i];
            t4 += b[i][j] * b[j][i];
        }
    }
    tr[0] = t1; tr[1] = t2; tr[2] = t3; tr[3] = t4;
}

// ---------------- kernel 1: filter traces * coeff -> wS tables ----------------
__global__ void prep_filters(const float* __restrict__ W_adj_tri,
                             const float* __restrict__ W_adj_cl4,
                             const float* __restrict__ length_gates,
                             float* __restrict__ wS_tri,   // [L_][F_]
                             float* __restrict__ wS_cl4) { // [L_][F_]
    const int t = threadIdx.x;
    if (t >= 2 * F_) return;
    float coeff[L_];
    float g = GAMMA;
    #pragma unroll
    for (int l = 0; l < L_; ++l) {
        coeff[l] = g * softplus_f(length_gates[l]);
        g *= GAMMA;
    }
    float tr[4];
    if (t < F_) {
        const int f = t;
        const float* w = W_adj_tri + f * 9;
        float a[3][3];
        #pragma unroll
        for (int i = 0; i < 3; ++i)
            #pragma unroll
            for (int j = 0; j < 3; ++j)
                a[i][j] = (i == j) ? 0.f
                                   : 0.5f * (softplus_f(w[i*3+j]) + softplus_f(w[j*3+i]));
        row_norm<3>(a);
        power_traces<3>(a, tr);
        #pragma unroll
        for (int l = 0; l < L_; ++l) wS_tri[l * F_ + f] = coeff[l] * tr[l];
    } else {
        const int f = t - F_;
        const float* w = W_adj_cl4 + f * 16;
        float a[4][4];
        #pragma unroll
        for (int i = 0; i < 4; ++i)
            #pragma unroll
            for (int j = 0; j < 4; ++j)
                a[i][j] = (i == j) ? 0.f
                                   : 0.5f * (softplus_f(w[i*4+j]) + softplus_f(w[j*4+i]));
        row_norm<4>(a);
        power_traces<4>(a, tr);
        #pragma unroll
        for (int l = 0; l < L_; ++l) wS_cl4[l * F_ + f] = coeff[l] * tr[l];
    }
}

// ---------------- kernel 2: bf16 transposed / padded weights ----------------
__global__ void convert_weights(const float* __restrict__ W1,   // [IN_CH][HID]
                                const float* __restrict__ W2,   // [HID][D_OUT]
                                __bf16* __restrict__ W1T,       // [HID][K1_PAD]
                                __bf16* __restrict__ W2T) {     // [D_OUT][HID]
    const int idx = blockIdx.x * blockDim.x + threadIdx.x;
    const int n1 = HID * K1_PAD;
    const int n2 = D_OUT * HID;
    if (idx < n1) {
        const int n = idx / K1_PAD, k = idx % K1_PAD;
        W1T[idx] = (k < IN_CH) ? (__bf16)W1[k * HID + n] : (__bf16)0.f;
    } else if (idx < n1 + n2) {
        const int j = idx - n1;
        const int n = j / HID, k = j % HID;
        W2T[j] = (__bf16)W2[k * D_OUT + n];
    }
}

// ---------------- kernel 3: Phi = [0 | 0 | H_in] ----------------
__global__ void init_phi(const float* __restrict__ H_in, float* __restrict__ Phi) {
    const long idx = (long)blockIdx.x * blockDim.x + threadIdx.x;
    if (idx >= (long)N_NODES * IN_CH) return;
    const int i = (int)(idx / IN_CH), c = (int)(idx % IN_CH);
    Phi[idx] = (c < 2 * F_) ? 0.f : H_in[(long)i * D_IN + (c - 2 * F_)];
}

// ---------------- kernel 4/5: motif response scatter ----------------
template <int S>
__global__ void motif_kernel(const float* __restrict__ A,
                             const int* __restrict__ nodes,   // [M][S]
                             const float* __restrict__ wS,    // [L_][F_]
                             float* __restrict__ Phi,         // [N][IN_CH]
                             int M, int col_off) {
    const int m = blockIdx.x * blockDim.x + threadIdx.x;
    if (m >= M) return;
    int nn[S];
    #pragma unroll
    for (int i = 0; i < S; ++i) nn[i] = nodes[m * S + i];
    float a[S][S];
    #pragma unroll
    for (int i = 0; i < S; ++i) {
        const float* row = A + (size_t)nn[i] * N_NODES;
        #pragma unroll
        for (int j = 0; j < S; ++j) a[i][j] = row[nn[j]];
    }
    row_norm<S>(a);
    float tr[4];
    power_traces<S>(a, tr);

    float sims[F_], mx = -3.4e38f;
    #pragma unroll
    for (int f = 0; f < F_; ++f) {
        float s = 0.f;
        #pragma unroll
        for (int l = 0; l < L_; ++l) s += tr[l] * wS[l * F_ + f];
        sims[f] = s;                      // TAU == 1.0
        mx = fmaxf(mx, s);
    }
    float Z = 0.f, e[F_];
    #pragma unroll
    for (int f = 0; f < F_; ++f) { e[f] = expf(sims[f] - mx); Z += e[f]; }
    const float invZ = 1.f / Z, invS = 1.f / (float)S;
    float contrib[F_];
    #pragma unroll
    for (int f = 0; f < F_; ++f) contrib[f] = e[f] * invZ * sims[f] * invS;

    #pragma unroll
    for (int i = 0; i < S; ++i) {
        float* dst = Phi + (size_t)nn[i] * IN_CH + col_off;
        #pragma unroll
        for (int f = 0; f < F_; ++f) atomicAdd(dst + f, contrib[f]);
    }
}

// ---------------- WMMA fragment K mapping (ISA 7.12.2, 16-bit A layout) ----------------
// element e (0..15) of v16bf, lane half selects K octet:
//   k = (v>=4 ? 16:0) + (lane>=16 ? 8:0) + (v&3)*2 + (e&1),  v = e>>1
__device__ __forceinline__ int wmma_k(int e, int khalf) {
    const int v = e >> 1;
    return ((v & 4) << 2) + khalf + ((v & 3) << 1) + (e & 1);
}

// ---------------- kernel 6: fused LayerNorm + MLP via WMMA ----------------
__global__ __launch_bounds__(256)
void mlp_kernel(const float* __restrict__ Phi,
                const float* __restrict__ ln_g, const float* __restrict__ ln_b,
                const __bf16* __restrict__ W1T,  // [HID][K1_PAD]
                const float* __restrict__ b1,
                const __bf16* __restrict__ W2T,  // [D_OUT][HID]
                const float* __restrict__ b2,
                float* __restrict__ out) {
    __shared__ __bf16 smem[ROWS_PER_BLOCK * K1_PAD];   // 36 KB; reused for h[64][256]

    const int tid = threadIdx.x;
    const int rowBase = blockIdx.x * ROWS_PER_BLOCK;

    // ---- Phase A: LayerNorm (one thread per row) -> bf16 LDS [64][288]
    if (tid < ROWS_PER_BLOCK) {
        const float* prow = Phi + (size_t)(rowBase + tid) * IN_CH;
        float s = 0.f, s2 = 0.f;
        #pragma unroll 4
        for (int c = 0; c < IN_CH; ++c) { float x = prow[c]; s += x; s2 += x * x; }
        const float mu = s * (1.f / IN_CH);
        const float var = s2 * (1.f / IN_CH) - mu * mu;
        const float rstd = rsqrtf(var + 1e-5f);
        for (int c = 0; c < IN_CH; ++c) {
            float x = (prow[c] - mu) * rstd * ln_g[c] + ln_b[c];
            smem[tid * K1_PAD + c] = (__bf16)x;
        }
        #pragma unroll
        for (int c = IN_CH; c < K1_PAD; ++c) smem[tid * K1_PAD + c] = (__bf16)0.f;
    }
    __syncthreads();

    const int lane  = tid & 31;
    const int wave  = tid >> 5;
    const int mtile = wave >> 1;           // 0..3  (16-row tile)
    const int nhalf = (wave & 1) * 128;    // 0 or 128
    const int lmod  = lane & 15;
    const int khalf = (lane & 16) ? 8 : 0;
    const int mrow  = mtile * 16 + lmod;

    v8f acc[8];
    #pragma unroll
    for (int nt = 0; nt < 8; ++nt) acc[nt] = (v8f){0.f,0.f,0.f,0.f,0.f,0.f,0.f,0.f};

    // ---- Phase B: GEMM1  h = lnPhi(64x288) @ W1(288x256)
    #pragma unroll
    for (int ks = 0; ks < K1_PAD / 32; ++ks) {
        const int kbase = ks * 32;
        v16bf afrag;
        #pragma unroll
        for (int e = 0; e < 16; ++e)
            afrag[e] = smem[mrow * K1_PAD + kbase + wmma_k(e, khalf)];
        #pragma unroll
        for (int nt = 0; nt < 8; ++nt) {
            const int ncol = nhalf + nt * 16 + lmod;
            const __bf16* bp = W1T + (size_t)ncol * K1_PAD + kbase;
            v16bf bfrag;
            #pragma unroll
            for (int e = 0; e < 16; ++e) bfrag[e] = bp[wmma_k(e, khalf)];
            acc[nt] = __builtin_amdgcn_wmma_f32_16x16x32_bf16(
                false, afrag, false, bfrag, (short)0, acc[nt], false, false);
        }
    }
    __syncthreads();   // done reading LN data from smem

    // ---- Phase C: bias + exact gelu -> smem as h[64][256] bf16
    #pragma unroll
    for (int nt = 0; nt < 8; ++nt) {
        const int ncol = nhalf + nt * 16 + lmod;
        const float bias = b1[ncol];
        #pragma unroll
        for (int v = 0; v < 8; ++v) {
            const int m = mtile * 16 + v + khalf;     // C/D layout: M = v (+8 upper lanes)
            const float x = acc[nt][v] + bias;
            const float g = 0.5f * x * (1.f + erff(x * 0.70710678118f));
            smem[m * HID + ncol] = (__bf16)g;
        }
    }
    __syncthreads();

    // ---- Phase D: GEMM2  out = h(64x256) @ W2(256x256)
    #pragma unroll
    for (int nt = 0; nt < 8; ++nt) acc[nt] = (v8f){0.f,0.f,0.f,0.f,0.f,0.f,0.f,0.f};
    #pragma unroll
    for (int ks = 0; ks < HID / 32; ++ks) {
        const int kbase = ks * 32;
        v16bf afrag;
        #pragma unroll
        for (int e = 0; e < 16; ++e)
            afrag[e] = smem[mrow * HID + kbase + wmma_k(e, khalf)];
        #pragma unroll
        for (int nt = 0; nt < 8; ++nt) {
            const int ncol = nhalf + nt * 16 + lmod;
            const __bf16* bp = W2T + (size_t)ncol * HID + kbase;
            v16bf bfrag;
            #pragma unroll
            for (int e = 0; e < 16; ++e) bfrag[e] = bp[wmma_k(e, khalf)];
            acc[nt] = __builtin_amdgcn_wmma_f32_16x16x32_bf16(
                false, afrag, false, bfrag, (short)0, acc[nt], false, false);
        }
    }

    // ---- Phase E: bias + store
    #pragma unroll
    for (int nt = 0; nt < 8; ++nt) {
        const int ncol = nhalf + nt * 16 + lmod;
        const float bias = b2[ncol];
        #pragma unroll
        for (int v = 0; v < 8; ++v) {
            const int m = mtile * 16 + v + khalf;
            out[(size_t)(rowBase + m) * D_OUT + ncol] = acc[nt][v] + bias;
        }
    }
}

// ---------------- launch ----------------
extern "C" void kernel_launch(void* const* d_in, const int* in_sizes, int n_in,
                              void* d_out, int out_size, void* d_ws, size_t ws_size,
                              hipStream_t stream) {
    const float* A_full   = (const float*)d_in[0];
    const float* H_in     = (const float*)d_in[1];
    const float* Wadj_tri = (const float*)d_in[2];
    const float* Wadj_cl4 = (const float*)d_in[3];
    const float* gates    = (const float*)d_in[4];
    const float* ln_g     = (const float*)d_in[5];
    const float* ln_b     = (const float*)d_in[6];
    const float* W1       = (const float*)d_in[7];
    const float* b1       = (const float*)d_in[8];
    const float* W2       = (const float*)d_in[9];
    const float* b2       = (const float*)d_in[10];
    const int*   nodes_tri= (const int*)d_in[11];
    const int*   nodes_cl4= (const int*)d_in[12];
    float* out = (float*)d_out;

    // workspace layout
    float* wS_tri = (float*)d_ws;                               // 32 f
    float* wS_cl4 = wS_tri + L_ * F_;                           // 32 f
    float* Phi    = wS_cl4 + L_ * F_;                           // N*272 f
    __bf16* W1T   = (__bf16*)(Phi + (size_t)N_NODES * IN_CH);   // 256*288 bf16
    __bf16* W2T   = W1T + HID * K1_PAD;                         // 256*256 bf16

    prep_filters<<<1, 32, 0, stream>>>(Wadj_tri, Wadj_cl4, gates, wS_tri, wS_cl4);

    {
        const int total = HID * K1_PAD + D_OUT * HID;
        convert_weights<<<(total + 255) / 256, 256, 0, stream>>>(W1, W2, W1T, W2T);
    }
    {
        const long total = (long)N_NODES * IN_CH;
        init_phi<<<(int)((total + 255) / 256), 256, 0, stream>>>(H_in, Phi);
    }

    motif_kernel<3><<<(M_TRI + 255) / 256, 256, 0, stream>>>(
        A_full, nodes_tri, wS_tri, Phi, M_TRI, 0);
    motif_kernel<4><<<(M_CL4 + 255) / 256, 256, 0, stream>>>(
        A_full, nodes_cl4, wS_cl4, Phi, M_CL4, F_);

    mlp_kernel<<<N_NODES / ROWS_PER_BLOCK, 256, 0, stream>>>(
        Phi, ln_g, ln_b, W1T, b1, W2T, b2, out);
}
